// SwinTransformerBlock_11132555231835
// MI455X (gfx1250) — matrix operand order repaired and checked by reading, not compiled, not run on hardware
//
#include <hip/hip_runtime.h>
#include <hip/hip_bf16.h>

// ---------------------------------------------------------------------------
// Swin Transformer block for MI455X (gfx1250, wave32, WMMA bf16).
// B=64 H=W=56 C=192 NH=6 WS=7 SS=3 N=49 NW=64 HD=32
// ---------------------------------------------------------------------------

typedef __bf16 bf16_t;
typedef __bf16  v16bf __attribute__((ext_vector_type(16)));
typedef float   v8f   __attribute__((ext_vector_type(8)));

#define BATCH 64
#define HH    56
#define WW_   56
#define CC    192
#define NHD   6
#define WS7   7
#define SS3   3
#define NTOK  49
#define NWIN  64
#define HD32  32
#define MROWS (BATCH * NWIN * NTOK)   // 200704, divisible by 16
#define C3    576
#define CH    768
#define SCALE 0.17677669529663687f    // 32^-0.5

union Frag16 { uint4 u[2]; v16bf v; };

static __device__ __forceinline__ v8f zero_v8f() {
  v8f z;
#pragma unroll
  for (int i = 0; i < 8; ++i) z[i] = 0.0f;
  return z;
}

static __device__ __forceinline__ v8f wmma_bf16(v16bf a, v16bf b, v8f c) {
  return __builtin_amdgcn_wmma_f32_16x16x32_bf16(false, a, false, b, (short)0, c,
                                                 false, false);
}

// A fragment (16x32, bf16), tile stored row-major [16][ld] at `base`.
// Layout: lane m=L&15, element i holds K=(L>>4)*16+i (same K order as load_B)
// => one contiguous 32-byte chunk per lane.
static __device__ __forceinline__ v16bf load_A(const bf16_t* base, int ld, int lane) {
  int m = lane & 15, kb = (lane >> 4) * 16;
  const bf16_t* row = base + (long)m * ld;
  Frag16 f;
  f.u[0] = *(const uint4*)(row + kb);
  f.u[1] = *(const uint4*)(row + kb + 8);
  return f.v;
}

// B fragment (32x16, bf16) where B[k][n] = W[n][k], W row-major [16 n][ld k].
// Layout (same K order as load_A): lane n=L&15, element i holds K=(L>>4)*16+i => one contiguous
// 32-byte chunk per lane.
static __device__ __forceinline__ v16bf load_B(const bf16_t* wbase, int ld, int lane) {
  int n = lane & 15, kb = (lane >> 4) * 16;
  const bf16_t* row = wbase + (long)n * ld;
  Frag16 f;
  f.u[0] = *(const uint4*)(row + kb);
  f.u[1] = *(const uint4*)(row + kb + 8);
  return f.v;
}

// B fragment where B[k][n] = V[k][n] (V row-major [k][ld n]): strided gather.
static __device__ __forceinline__ v16bf load_Bt(const bf16_t* v, int ld, int koff,
                                                int noff, int lane) {
  int n = noff + (lane & 15), kb = koff + ((lane >> 4) * 16);
  v16bf b;
#pragma unroll
  for (int i = 0; i < 16; ++i) b[i] = v[(long)(kb + i) * ld + n];
  return b;
}

// row r of the shifted/windowed layout -> flat pixel index in [B,H,W]
static __device__ __forceinline__ int pixel_of(int r) {
  int w = r / NTOK, n = r - w * NTOK;
  int b = w >> 6, wi = w & 63;
  int wh = wi >> 3, wwi = wi & 7;
  int i = n / WS7, j = n - i * WS7;
  int ph = (wh * WS7 + i + SS3) % HH;
  int pw = (wwi * WS7 + j + SS3) % WW_;
  return (b * HH + ph) * WW_ + pw;
}

// ---------------------------------------------------------------------------
// K0: convert all weights fp32 -> bf16 (stay hot in the 192MB L2 afterwards)
// ---------------------------------------------------------------------------
__global__ void k_cvt_weights(const float* __restrict__ qw, const float* __restrict__ pw,
                              const float* __restrict__ f1, const float* __restrict__ f2,
                              bf16_t* wq, bf16_t* wp, bf16_t* w1, bf16_t* w2) {
  int i = blockIdx.x * 256 + threadIdx.x;
  if (i < C3 * CC) wq[i] = (bf16_t)qw[i];
  if (i < CC * CC) wp[i] = (bf16_t)pw[i];
  if (i < CH * CC) { w1[i] = (bf16_t)f1[i]; w2[i] = (bf16_t)f2[i]; }
}

// ---------------------------------------------------------------------------
// K1: LN1 + cyclic shift + window partition -> xw bf16 [MROWS, 192]
// one wave per token row; lane handles channels lane+32k
// ---------------------------------------------------------------------------
__global__ void k_prep_ln1(const float* __restrict__ x, const float* __restrict__ g,
                           const float* __restrict__ bt, bf16_t* __restrict__ xw) {
  int lane = threadIdx.x & 31, wv = threadIdx.x >> 5;
  int r = blockIdx.x * 4 + wv;
  const float* src = x + (long)pixel_of(r) * CC;
  float vals[6], sum = 0.f, sq = 0.f;
#pragma unroll
  for (int k = 0; k < 6; ++k) {
    float f = src[lane + 32 * k];
    vals[k] = f; sum += f; sq += f * f;
  }
#pragma unroll
  for (int off = 16; off >= 1; off >>= 1) {
    sum += __shfl_xor(sum, off, 32);
    sq  += __shfl_xor(sq,  off, 32);
  }
  float mean = sum * (1.0f / CC);
  float var  = sq * (1.0f / CC) - mean * mean;
  float rstd = rsqrtf(var + 1e-5f);
  bf16_t* dst = xw + (long)r * CC;
#pragma unroll
  for (int k = 0; k < 6; ++k) {
    int c = lane + 32 * k;
    dst[c] = (bf16_t)((vals[k] - mean) * rstd * g[c] + bt[c]);
  }
}

// ---------------------------------------------------------------------------
// K2: QKV GEMM  qkv[r,d] = sum_c xw[r,c]*qkv_w[d,c] + qkv_b[d], bf16 out.
// block = 128 (4 waves); block owns one 16-row M tile; each wave 9 N tiles.
// B loads are left to the compiler's scheduler (explicit double-buffering
// caused scratch spills); a prefetch hint covers the next tile's weights.
// ---------------------------------------------------------------------------
__global__ void k_qkv_gemm(const bf16_t* __restrict__ xw, const bf16_t* __restrict__ wq,
                           const float* __restrict__ qb, bf16_t* __restrict__ qkv) {
  int lane = threadIdx.x & 31, wv = threadIdx.x >> 5;
  int half = lane >> 4, nloc = lane & 15;
  long mt = blockIdx.x;
  const bf16_t* Abase = xw + mt * 16 * CC;
  v16bf a[6];
#pragma unroll
  for (int ks = 0; ks < 6; ++ks) a[ks] = load_A(Abase + ks * 32, CC, lane);
  for (int t = 0; t < 9; ++t) {
    int nt = wv * 9 + t;                       // 0..35
    const bf16_t* Bbase = wq + (long)nt * 16 * CC;
    if (t < 8) __builtin_prefetch(Bbase + (long)16 * CC, 0, 1);  // next tile
    v8f acc = zero_v8f();
#pragma unroll
    for (int ks = 0; ks < 6; ++ks)
      acc = wmma_bf16(a[ks], load_B(Bbase + ks * 32, CC, lane), acc);
    int n = nt * 16 + nloc;
    float bias = qb[n];
#pragma unroll
    for (int v = 0; v < 8; ++v) {
      long r = mt * 16 + v + 8 * half;
      qkv[r * C3 + n] = (bf16_t)(acc[v] + bias);
    }
  }
}

// ---------------------------------------------------------------------------
// K3: windowed attention.  One wave per (window, head); 4 waves per block.
// S (64x64 padded) lives in 128 accumulator VGPRs; softmax in registers via
// shfl_xor; V staged into LDS via GLOBAL_LOAD_ASYNC_TO_LDS (ASYNCcnt path),
// P(bf16) written to LDS for the P*V WMMAs.
// ---------------------------------------------------------------------------
__global__ void k_attn(const bf16_t* __restrict__ qkv, const float* __restrict__ rpb,
                       const float* __restrict__ mask, bf16_t* __restrict__ ao) {
  __shared__ bf16_t smem[4 * (64 * 32 + 64 * 64)];   // per wave: V 4KB + P 8KB
  int lane = threadIdx.x & 31, wv = threadIdx.x >> 5;
  int half = lane >> 4, nloc = lane & 15;
  bf16_t* V = smem + wv * (64 * 32 + 64 * 64);
  bf16_t* P = V + 64 * 32;

  int pair = blockIdx.x * 4 + wv;
  int w = pair / NHD, h = pair - (pair / NHD) * NHD;
  int wi = w & 63;
  long qbase = (long)w * NTOK * C3;

  // ---- stage V rows 0..48 via async DMA global->LDS (64B per row) ----
  for (int rr = lane; rr < 64; rr += 32) {
    if (rr < NTOK) {
      unsigned long long ga =
          (unsigned long long)(uintptr_t)(qkv + qbase + (long)rr * C3 + 2 * CC + h * HD32);
      // low 32 bits of the generic shared-pointer == wave-relative LDS offset
      unsigned ldsoff = (unsigned)(uintptr_t)(V + rr * 32);
      asm volatile("global_load_async_to_lds_b128 %0, %1, off"
                   :: "v"(ldsoff), "v"(ga) : "memory");
      asm volatile("global_load_async_to_lds_b128 %0, %1, off offset:16"
                   :: "v"(ldsoff), "v"(ga) : "memory");
      asm volatile("global_load_async_to_lds_b128 %0, %1, off offset:32"
                   :: "v"(ldsoff), "v"(ga) : "memory");
      asm volatile("global_load_async_to_lds_b128 %0, %1, off offset:48"
                   :: "v"(ldsoff), "v"(ga) : "memory");
    }
  }
  // zero-fill pad rows 49..63 (15 rows * 64B = 240 dwords)
  {
    unsigned* zp = (unsigned*)(V + NTOK * 32);
    for (int i = lane; i < 240; i += 32) zp[i] = 0u;
  }

  // ---- Q fragments + S = Q*K^T while the DMA runs ----
  v16bf aq[4];
#pragma unroll
  for (int mt = 0; mt < 4; ++mt)
    aq[mt] = load_A(qkv + qbase + (long)mt * 16 * C3 + h * HD32, C3, lane);

  v8f s[4][4];
#pragma unroll
  for (int nt = 0; nt < 4; ++nt) {
    v16bf bk = load_B(qkv + qbase + (long)nt * 16 * C3 + CC + h * HD32, C3, lane);
#pragma unroll
    for (int mt = 0; mt < 4; ++mt) s[mt][nt] = wmma_bf16(aq[mt], bk, zero_v8f());
  }

  asm volatile("s_wait_asynccnt 0x0" ::: "memory");
  __syncthreads();

  // scale + rel-pos bias + shift mask + softmax (per row), write P to LDS
#pragma unroll
  for (int mt = 0; mt < 4; ++mt) {
#pragma unroll
    for (int v = 0; v < 8; ++v) {
      int m = mt * 16 + v + 8 * half;
#pragma unroll
      for (int nt = 0; nt < 4; ++nt) {
        int n = nt * 16 + nloc;
        float val;
        if (n < NTOK && m < NTOK) {
          int mi = m / WS7, mj = m - mi * WS7;
          int ni = n / WS7, nj = n - ni * WS7;
          int idx = (mi - ni + WS7 - 1) * (2 * WS7 - 1) + (mj - nj + WS7 - 1);
          val = s[mt][nt][v] * SCALE + rpb[idx * NHD + h]
              + mask[(long)wi * NTOK * NTOK + m * NTOK + n];
        } else {
          val = (n < NTOK) ? 0.0f : -1e30f;   // pad cols -> -inf, pad rows dummy
        }
        s[mt][nt][v] = val;
      }
      float rm = fmaxf(fmaxf(s[mt][0][v], s[mt][1][v]),
                       fmaxf(s[mt][2][v], s[mt][3][v]));
#pragma unroll
      for (int off = 8; off >= 1; off >>= 1) rm = fmaxf(rm, __shfl_xor(rm, off, 32));
      float rs = 0.f;
#pragma unroll
      for (int nt = 0; nt < 4; ++nt) {
        float p = __expf(s[mt][nt][v] - rm);
        s[mt][nt][v] = p; rs += p;
      }
#pragma unroll
      for (int off = 8; off >= 1; off >>= 1) rs += __shfl_xor(rs, off, 32);
      float inv = 1.0f / rs;
#pragma unroll
      for (int nt = 0; nt < 4; ++nt)
        P[m * 64 + nt * 16 + nloc] = (bf16_t)(s[mt][nt][v] * inv);
    }
  }
  __syncthreads();

  // O = P * V  (16 WMMAs), write rows < 49 as bf16 into ao[r, h*32+hd]
  v16bf bv[2][2];
#pragma unroll
  for (int kt = 0; kt < 2; ++kt)
#pragma unroll
    for (int nto = 0; nto < 2; ++nto)
      bv[kt][nto] = load_Bt(V, 32, kt * 32, nto * 16, lane);

  long obase = (long)w * NTOK * CC + h * HD32;
#pragma unroll
  for (int mt = 0; mt < 4; ++mt) {
    v16bf ap0 = load_A(P + mt * 16 * 64, 64, lane);
    v16bf ap1 = load_A(P + mt * 16 * 64 + 32, 64, lane);
#pragma unroll
    for (int nto = 0; nto < 2; ++nto) {
      v8f o = wmma_bf16(ap0, bv[0][nto], zero_v8f());
      o = wmma_bf16(ap1, bv[1][nto], o);
#pragma unroll
      for (int v = 0; v < 8; ++v) {
        int m = mt * 16 + v + 8 * half;
        if (m < NTOK)
          ao[obase + (long)m * CC + nto * 16 + nloc] = (bf16_t)o[v];
      }
    }
  }
}

// ---------------------------------------------------------------------------
// K4: proj GEMM + window-reverse + un-shift + residual -> x1 (fp32)
// ---------------------------------------------------------------------------
__global__ void k_proj_res(const bf16_t* __restrict__ ao, const bf16_t* __restrict__ wp,
                           const float* __restrict__ pb, const float* __restrict__ x,
                           float* __restrict__ x1) {
  int lane = threadIdx.x & 31, wv = threadIdx.x >> 5;
  int half = lane >> 4, nloc = lane & 15;
  long mt = blockIdx.x;
  const bf16_t* Abase = ao + mt * 16 * CC;
  v16bf a[6];
#pragma unroll
  for (int ks = 0; ks < 6; ++ks) a[ks] = load_A(Abase + ks * 32, CC, lane);
  long pix[8];
#pragma unroll
  for (int v = 0; v < 8; ++v) pix[v] = pixel_of((int)(mt * 16 + v + 8 * half));
  for (int t = 0; t < 3; ++t) {
    int nt = wv * 3 + t;                       // 0..11
    const bf16_t* Bbase = wp + (long)nt * 16 * CC;
    v8f acc = zero_v8f();
#pragma unroll
    for (int ks = 0; ks < 6; ++ks)
      acc = wmma_bf16(a[ks], load_B(Bbase + ks * 32, CC, lane), acc);
    int n = nt * 16 + nloc;
    float bias = pb[n];
#pragma unroll
    for (int v = 0; v < 8; ++v)
      x1[pix[v] * CC + n] = x[pix[v] * CC + n] + acc[v] + bias;
  }
}

// ---------------------------------------------------------------------------
// K5: fused LN2 -> fc1 -> GELU(exact) -> fc2 -> residual.
// block = 128 (4 waves) owns 16 token rows; hidden [16,768] bf16 kept in LDS.
// ---------------------------------------------------------------------------
__global__ void k_mlp(const float* __restrict__ x1, const bf16_t* __restrict__ w1,
                      const float* __restrict__ b1, const bf16_t* __restrict__ w2,
                      const float* __restrict__ b2, const float* __restrict__ g2,
                      const float* __restrict__ be2, float* __restrict__ out) {
  __shared__ bf16_t xn[16 * CC];     // 6 KB
  __shared__ bf16_t hh[16 * CH];     // 24 KB
  int tid = threadIdx.x;
  int lane = tid & 31, wv = tid >> 5;
  int half = lane >> 4, nloc = lane & 15;
  long r0 = (long)blockIdx.x * 16;

  // --- LN2: 8 threads per row ---
  int gr = tid >> 3, sub = tid & 7;
  const float* row = x1 + (r0 + gr) * CC;
  float vals[24], sum = 0.f, sq = 0.f;
#pragma unroll
  for (int k = 0; k < 24; ++k) {
    float f = row[sub + 8 * k];
    vals[k] = f; sum += f; sq += f * f;
  }
#pragma unroll
  for (int off = 4; off >= 1; off >>= 1) {
    sum += __shfl_xor(sum, off, 32);
    sq  += __shfl_xor(sq,  off, 32);
  }
  float mean = sum * (1.0f / CC);
  float var  = sq * (1.0f / CC) - mean * mean;
  float rstd = rsqrtf(var + 1e-5f);
#pragma unroll
  for (int k = 0; k < 24; ++k) {
    int c = sub + 8 * k;
    xn[gr * CC + c] = (bf16_t)((vals[k] - mean) * rstd * g2[c] + be2[c]);
  }
  __syncthreads();

  // --- fc1 + GELU into LDS hidden tile ---
  v16bf a[6];
#pragma unroll
  for (int ks = 0; ks < 6; ++ks) a[ks] = load_A(xn + ks * 32, CC, lane);
  for (int t = 0; t < 12; ++t) {
    int nt = wv * 12 + t;                      // 0..47
    const bf16_t* Bbase = w1 + (long)nt * 16 * CC;
    if (t < 11) __builtin_prefetch(Bbase + (long)16 * CC, 0, 1);  // next tile
    v8f acc = zero_v8f();
#pragma unroll
    for (int ks = 0; ks < 6; ++ks)
      acc = wmma_bf16(a[ks], load_B(Bbase + ks * 32, CC, lane), acc);
    int n = nt * 16 + nloc;
    float bias = b1[n];
#pragma unroll
    for (int v = 0; v < 8; ++v) {
      float z = acc[v] + bias;
      float ge = 0.5f * z * (1.0f + erff(z * 0.70710678118654752f));
      hh[(v + 8 * half) * CH + n] = (bf16_t)ge;
    }
  }
  __syncthreads();

  // --- fc2 + residual ---
  for (int t = 0; t < 3; ++t) {
    int nt = wv * 3 + t;                       // 0..11
    const bf16_t* Bbase = w2 + (long)nt * 16 * CH;
    __builtin_prefetch(Bbase, 0, 1);
    v8f acc = zero_v8f();
#pragma unroll
    for (int ks = 0; ks < 24; ++ks)
      acc = wmma_bf16(load_A(hh + ks * 32, CH, lane),
                      load_B(Bbase + ks * 32, CH, lane), acc);
    int n = nt * 16 + nloc;
    float bias = b2[n];
#pragma unroll
    for (int v = 0; v < 8; ++v) {
      long r = r0 + v + 8 * half;
      out[r * CC + n] = x1[r * CC + n] + acc[v] + bias;
    }
  }
}

// ---------------------------------------------------------------------------
extern "C" void kernel_launch(void* const* d_in, const int* in_sizes, int n_in,
                              void* d_out, int out_size, void* d_ws, size_t ws_size,
                              hipStream_t stream) {
  const float* x     = (const float*)d_in[0];
  const float* mask  = (const float*)d_in[1];
  const float* n1g   = (const float*)d_in[2];
  const float* n1b   = (const float*)d_in[3];
  const float* qkvw  = (const float*)d_in[4];
  const float* qkvb  = (const float*)d_in[5];
  const float* rpb   = (const float*)d_in[6];
  const float* projw = (const float*)d_in[7];
  const float* projb = (const float*)d_in[8];
  const float* n2g   = (const float*)d_in[9];
  const float* n2b   = (const float*)d_in[10];
  const float* f1w   = (const float*)d_in[11];
  const float* f1b   = (const float*)d_in[12];
  const float* f2w   = (const float*)d_in[13];
  const float* f2b   = (const float*)d_in[14];
  (void)in_sizes; (void)n_in; (void)out_size; (void)ws_size;

  char* ws = (char*)d_ws;
  size_t o = 0;
  auto alloc = [&](size_t bytes) -> char* {
    char* p = ws + o;
    o = (o + bytes + 255) & ~(size_t)255;
    return p;
  };
  bf16_t* wq  = (bf16_t*)alloc((size_t)C3 * CC * 2);
  bf16_t* wp  = (bf16_t*)alloc((size_t)CC * CC * 2);
  bf16_t* w1  = (bf16_t*)alloc((size_t)CH * CC * 2);
  bf16_t* w2  = (bf16_t*)alloc((size_t)CC * CH * 2);
  bf16_t* xw  = (bf16_t*)alloc((size_t)MROWS * CC * 2);        // 77 MB
  bf16_t* qkv = (bf16_t*)alloc((size_t)MROWS * C3 * 2 + 32768);// 231 MB (+pad slack)
  bf16_t* ao  = (bf16_t*)alloc((size_t)MROWS * CC * 2);        // 77 MB
  float*  x1  = (float*) alloc((size_t)MROWS * CC * 4);        // 154 MB

  k_cvt_weights<<<(CH * CC + 255) / 256, 256, 0, stream>>>(qkvw, projw, f1w, f2w,
                                                           wq, wp, w1, w2);
  k_prep_ln1<<<MROWS / 4, 128, 0, stream>>>(x, n1g, n1b, xw);
  k_qkv_gemm<<<MROWS / 16, 128, 0, stream>>>(xw, wq, qkvb, qkv);
  k_attn<<<(BATCH * NWIN * NHD) / 4, 128, 0, stream>>>(qkv, rpb, mask, ao);
  k_proj_res<<<MROWS / 16, 128, 0, stream>>>(ao, wp, projb, x, x1);
  k_mlp<<<MROWS / 16, 128, 0, stream>>>(x1, w1, f1b, w2, f2b, n2g, n2b,
                                        (float*)d_out);
}
